// LSTMSeq2Seq_67946382623039
// MI455X (gfx1250) — matrix operand order repaired and checked by reading, not compiled, not run on hardware
//
#include <hip/hip_runtime.h>

// LSTM seq2seq fused kernel for gfx1250 (MI455X).
// B=4096, T=512, IN=16, OUT=16, H=64, gates 4H=256.
// One wave32 owns one 16-row batch tile for the entire enc+dec recurrence.

#define T_STEPS 512
#define H_DIM   64
#define G_DIM   256     // 4*H
#define KP      96      // packed K: 64 (h) + 16 (x/prev) + 16 (zero pad)
#define IN_DIM  16

typedef _Float16 v16h __attribute__((ext_vector_type(16)));
typedef _Float16 v8h  __attribute__((ext_vector_type(8)));
typedef float    v8f  __attribute__((ext_vector_type(8)));

#define WMMA_F16(A, B, C) \
  __builtin_amdgcn_wmma_f32_16x16x32_f16(false, (A), false, (B), (short)0, (C), false, false)

__device__ __forceinline__ float sigm(float x) {
  return 1.0f / (1.0f + __expf(-x));
}
__device__ __forceinline__ float tanh_fast(float x) {
  float e = __expf(-2.0f * x);
  return (1.0f - e) / (1.0f + e);
}

// B-fragment (32x16 KxN slice, f16): lane n(0..15)=col n K0..7,16..23, lane n+16=col n K8..15,24..31.
// Weights stored [col][KP] in LDS -> each lane reads two 8-half runs (16B each).
__device__ __forceinline__ v16h lds_bfrag(const _Float16* W, int col, int ch, int hi) {
  v8h lo = *(const v8h*)(W + col * KP + ch * 32 + hi * 8);
  v8h hh = *(const v8h*)(W + col * KP + ch * 32 + 16 + hi * 8);
  return __builtin_shufflevector(lo, hh, 0, 1, 2, 3, 4, 5, 6, 7, 8, 9, 10, 11, 12, 13, 14, 15);
}
__device__ __forceinline__ v16h lds_bfrag_head(const _Float16* W, int col, int ch, int hi) {
  v8h lo = *(const v8h*)(W + col * H_DIM + ch * 32 + hi * 8);
  v8h hh = *(const v8h*)(W + col * H_DIM + ch * 32 + 16 + hi * 8);
  return __builtin_shufflevector(lo, hh, 0, 1, 2, 3, 4, 5, 6, 7, 8, 9, 10, 11, 12, 13, 14, 15);
}
// A-fragment (16x32 MxK slice) from row-major 16xH LDS h-buffer.
// lane<16: elems0..7=K(c*32+0..7), elems8..15=K(c*32+16..23); lane>=16: +8 shift.
__device__ __forceinline__ v16h lds_afrag(const _Float16* Hb, int row, int ch, int hi) {
  v8h lo = *(const v8h*)(Hb + row * H_DIM + ch * 32 + hi * 8);
  v8h hh = *(const v8h*)(Hb + row * H_DIM + ch * 32 + 16 + hi * 8);
  return __builtin_shufflevector(lo, hh, 0, 1, 2, 3, 4, 5, 6, 7, 8, 9, 10, 11, 12, 13, 14, 15);
}

// ---- prep kernel: pack weights to f16, K-concatenated [Whh | Wih | 0pad] per gate row ----
// ws layout (halves): [0, 24576) enc, [24576, 49152) dec, [49152, 50176) head.
__global__ void pack_weights_kernel(const float* __restrict__ enc_Wih,
                                    const float* __restrict__ enc_Whh,
                                    const float* __restrict__ dec_Wih,
                                    const float* __restrict__ dec_Whh,
                                    const float* __restrict__ head_W,
                                    _Float16* __restrict__ ws) {
  const int ENC = G_DIM * KP;
  int idx = blockIdx.x * 256 + threadIdx.x;
  if (idx < ENC) {
    int n = idx / KP, k = idx % KP;
    float v = (k < H_DIM) ? enc_Whh[n * H_DIM + k]
            : (k < H_DIM + IN_DIM ? enc_Wih[n * IN_DIM + (k - H_DIM)] : 0.0f);
    ws[idx] = (_Float16)v;
  } else if (idx < 2 * ENC) {
    int j = idx - ENC;
    int n = j / KP, k = j % KP;
    float v = (k < H_DIM) ? dec_Whh[n * H_DIM + k]
            : (k < H_DIM + IN_DIM ? dec_Wih[n * IN_DIM + (k - H_DIM)] : 0.0f);
    ws[idx] = (_Float16)v;
  } else if (idx < 2 * ENC + 16 * H_DIM) {
    ws[idx] = (_Float16)head_W[idx - 2 * ENC];
  }
}

// ---- main fused kernel: 64 blocks x 128 threads (4 waves), one wave = one 16-row tile ----
__launch_bounds__(128, 1)
__global__ void lstm_s2s_kernel(const float* __restrict__ X,
                                const float* __restrict__ enc_b,
                                const float* __restrict__ dec_b,
                                const float* __restrict__ head_b,
                                const _Float16* __restrict__ wpack,
                                float* __restrict__ out) {
  __shared__ __attribute__((aligned(32))) _Float16 sWenc[G_DIM * KP];     // 48 KB
  __shared__ __attribute__((aligned(32))) _Float16 sWdec[G_DIM * KP];     // 48 KB
  __shared__ __attribute__((aligned(32))) _Float16 sWhead[16 * H_DIM];    //  2 KB
  __shared__ __attribute__((aligned(32))) _Float16 sH[4][16 * H_DIM];     //  8 KB (per-wave h)
  __shared__ __attribute__((aligned(32))) _Float16 sPred[4][16 * 16];     //  2 KB (per-wave prev)

  const int tid = threadIdx.x;

  // Cooperative weight stage: global (packed f16) -> LDS.
  {
    const uint4* s0 = (const uint4*)(wpack);
    const uint4* s1 = (const uint4*)(wpack + G_DIM * KP);
    const uint4* s2 = (const uint4*)(wpack + 2 * G_DIM * KP);
    uint4* d0 = (uint4*)sWenc;
    uint4* d1 = (uint4*)sWdec;
    uint4* d2 = (uint4*)sWhead;
    for (int i = tid; i < (G_DIM * KP) / 8; i += 128) { d0[i] = s0[i]; d1[i] = s1[i]; }
    d2[tid & 127] = s2[tid & 127];
  }
  __syncthreads();

  const int wv   = tid >> 5;
  const int lane = tid & 31;
  const int row  = lane & 15;   // A-frag: M row; B/D frag: N col
  const int hi   = lane >> 4;
  const int b0   = (blockIdx.x * 4 + wv) * 16;

  _Float16* myH = sH[wv];
  _Float16* myP = sPred[wv];
  for (int i = lane; i < 16 * H_DIM; i += 32) myH[i] = (_Float16)0.0f;
  for (int i = lane; i < 16 * 16;   i += 32) myP[i] = (_Float16)0.0f;

  const v8f  z8  = {0.f, 0.f, 0.f, 0.f, 0.f, 0.f, 0.f, 0.f};
  const v16h z16 = {(_Float16)0, (_Float16)0, (_Float16)0, (_Float16)0,
                    (_Float16)0, (_Float16)0, (_Float16)0, (_Float16)0,
                    (_Float16)0, (_Float16)0, (_Float16)0, (_Float16)0,
                    (_Float16)0, (_Float16)0, (_Float16)0, (_Float16)0};

  // Biases live in registers: gate column = u*16 + row (same lane across i/f/g/o n-tiles).
  float bI[4], bF[4], bG[4], bO[4];
#pragma unroll
  for (int u = 0; u < 4; ++u) {
    int col = u * 16 + row;
    bI[u] = enc_b[col];
    bF[u] = enc_b[64 + col];
    bG[u] = enc_b[128 + col];
    bO[u] = enc_b[192 + col];
  }
  const float hb = head_b[row];

  // Cell state in registers: lane covers col u*16+row, rows (0..7)+hi*8.
  float cc[4][8];
#pragma unroll
  for (int u = 0; u < 4; ++u)
#pragma unroll
    for (int g = 0; g < 8; ++g) cc[u][g] = 0.0f;

  v16h a0 = z16, a1 = z16;  // h A-fragments (h starts at zero)

  // =================== encoder ===================
#pragma unroll 1
  for (int t = 0; t < T_STEPS; ++t) {
    // x_t A-fragment (K chunk 2: real K 0..15 = x, 16..31 = zero pad)
    const float* xp = X + ((size_t)(b0 + row) * T_STEPS + t) * IN_DIM + hi * 8;
    float4 x0 = *(const float4*)xp;
    float4 x1 = *(const float4*)(xp + 4);
    v16h a2 = z16;
    a2[0] = (_Float16)x0.x; a2[1] = (_Float16)x0.y; a2[2] = (_Float16)x0.z; a2[3] = (_Float16)x0.w;
    a2[4] = (_Float16)x1.x; a2[5] = (_Float16)x1.y; a2[6] = (_Float16)x1.z; a2[7] = (_Float16)x1.w;
    __builtin_prefetch(xp + 2 * IN_DIM, 0, 1);  // stream timestep t+2 of X

#pragma unroll
    for (int u = 0; u < 4; ++u) {
      v8f di = z8, df = z8, dg = z8, dq = z8;
#pragma unroll
      for (int ch = 0; ch < 3; ++ch) {
        v16h a = (ch == 0) ? a0 : (ch == 1 ? a1 : a2);
        di = WMMA_F16(a, lds_bfrag(sWenc, (u     ) * 16 + row, ch, hi), di);
        df = WMMA_F16(a, lds_bfrag(sWenc, (u +  4) * 16 + row, ch, hi), df);
        dg = WMMA_F16(a, lds_bfrag(sWenc, (u +  8) * 16 + row, ch, hi), dg);
        dq = WMMA_F16(a, lds_bfrag(sWenc, (u + 12) * 16 + row, ch, hi), dq);
      }
      // lane-local LSTM cell (i,f,g,o all in this lane's registers)
#pragma unroll
      for (int g = 0; g < 8; ++g) {
        float iv = sigm(di[g] + bI[u]);
        float fv = sigm(df[g] + bF[u]);
        float gv = tanh_fast(dg[g] + bG[u]);
        float ov = sigm(dq[g] + bO[u]);
        float cv = fv * cc[u][g] + iv * gv;
        cc[u][g] = cv;
        myH[(g + hi * 8) * H_DIM + u * 16 + row] = (_Float16)(ov * tanh_fast(cv));
      }
    }
    // D-layout (lane=col) -> A-layout (lane=row) via same-wave LDS round-trip
    a0 = lds_afrag(myH, row, 0, hi);
    a1 = lds_afrag(myH, row, 1, hi);
  }

  // =================== decoder ===================
#pragma unroll
  for (int u = 0; u < 4; ++u) {
    int col = u * 16 + row;
    bI[u] = dec_b[col];
    bF[u] = dec_b[64 + col];
    bG[u] = dec_b[128 + col];
    bO[u] = dec_b[192 + col];
  }

#pragma unroll 1
  for (int t = 0; t < T_STEPS; ++t) {
    // prev-pred A-fragment from LDS (16x16, K 16..31 zero)
    v8h pv = *(const v8h*)(myP + row * 16 + hi * 8);
    v16h a2 = z16;
#pragma unroll
    for (int j = 0; j < 8; ++j) a2[j] = pv[j];

#pragma unroll
    for (int u = 0; u < 4; ++u) {
      v8f di = z8, df = z8, dg = z8, dq = z8;
#pragma unroll
      for (int ch = 0; ch < 3; ++ch) {
        v16h a = (ch == 0) ? a0 : (ch == 1 ? a1 : a2);
        di = WMMA_F16(a, lds_bfrag(sWdec, (u     ) * 16 + row, ch, hi), di);
        df = WMMA_F16(a, lds_bfrag(sWdec, (u +  4) * 16 + row, ch, hi), df);
        dg = WMMA_F16(a, lds_bfrag(sWdec, (u +  8) * 16 + row, ch, hi), dg);
        dq = WMMA_F16(a, lds_bfrag(sWdec, (u + 12) * 16 + row, ch, hi), dq);
      }
#pragma unroll
      for (int g = 0; g < 8; ++g) {
        float iv = sigm(di[g] + bI[u]);
        float fv = sigm(df[g] + bF[u]);
        float gv = tanh_fast(dg[g] + bG[u]);
        float ov = sigm(dq[g] + bO[u]);
        float cv = fv * cc[u][g] + iv * gv;
        cc[u][g] = cv;
        myH[(g + hi * 8) * H_DIM + u * 16 + row] = (_Float16)(ov * tanh_fast(cv));
      }
    }
    a0 = lds_afrag(myH, row, 0, hi);
    a1 = lds_afrag(myH, row, 1, hi);

    // head: pred = h_t @ head_W^T + head_b  (M=16, K=64, N=16 -> 2 WMMAs)
    v8f dh = z8;
    dh = WMMA_F16(a0, lds_bfrag_head(sWhead, row, 0, hi), dh);
    dh = WMMA_F16(a1, lds_bfrag_head(sWhead, row, 1, hi), dh);
#pragma unroll
    for (int g = 0; g < 8; ++g) {
      float pr = dh[g] + hb;
      int r = g + hi * 8;
      out[((size_t)(b0 + r) * T_STEPS + t) * 16 + row] = pr;
      myP[r * 16 + row] = (_Float16)pr;  // feed back as prev for t+1
    }
  }
}

extern "C" void kernel_launch(void* const* d_in, const int* in_sizes, int n_in,
                              void* d_out, int out_size, void* d_ws, size_t ws_size,
                              hipStream_t stream) {
  (void)in_sizes; (void)n_in; (void)out_size; (void)ws_size;
  const float* X       = (const float*)d_in[0];
  const float* enc_Wih = (const float*)d_in[1];
  const float* enc_Whh = (const float*)d_in[2];
  const float* enc_b   = (const float*)d_in[3];
  const float* dec_Wih = (const float*)d_in[4];
  const float* dec_Whh = (const float*)d_in[5];
  const float* dec_b   = (const float*)d_in[6];
  const float* head_W  = (const float*)d_in[7];
  const float* head_b  = (const float*)d_in[8];
  float* out = (float*)d_out;
  _Float16* wpack = (_Float16*)d_ws;

  const int total_pack = 2 * G_DIM * KP + 16 * H_DIM;  // 50176 halves (~100 KB)
  pack_weights_kernel<<<(total_pack + 255) / 256, 256, 0, stream>>>(
      enc_Wih, enc_Whh, dec_Wih, dec_Whh, head_W, wpack);

  // 4096 batch rows / (4 waves * 16 rows) = 64 blocks of 128 threads.
  lstm_s2s_kernel<<<64, 128, 0, stream>>>(X, enc_b, dec_b, head_b, wpack, out);
}
